// Seq2SeqARDiffusion_87041807221330
// MI455X (gfx1250) — compile-verified, matrix operand order checked by reading, not run on hardware
//
#include <hip/hip_runtime.h>

typedef _Float16 half_t;
typedef __attribute__((ext_vector_type(16))) _Float16 v16h;
typedef __attribute__((ext_vector_type(8)))  _Float16 v8h;
typedef __attribute__((ext_vector_type(8)))  float    v8f;

// global-address-space views of the f16 weight workspace -> global_load_b128 (GVS form)
typedef const half_t __attribute__((address_space(1))) * gwptr;
typedef const v8h    __attribute__((address_space(1))) * gv8p;

#define T_IN   96
#define F_IN   8
#define EXO    6
#define EMBD   16
#define PRED   8
#define HID    128
#define TSTEPS 100
#define NB     1024

// workspace layout (in half elements), K padded to multiples of 32
#define OFF_W0   0        // [512][32]   (W_ih0, 24 -> 32 padded)
#define OFF_WH0  16384    // [512][128]  W_hh0
#define OFF_WI1  81920    // [512][128]  W_ih1
#define OFF_WH1  147456   // [512][128]  W_hh1
#define OFF_MW1  212992   // [128][192]  W1 (184 -> 192 padded)
#define OFF_MW2  237568   // [128][128]  W2
#define WS_HALFS 253952

// elements 0..7 = K k0..k0+7 ; elements 8..15 = K k0+16..k0+23 (ISA 16-bit A/B layout)
__device__ __forceinline__ v16h load_frag_s(const half_t* p) {   // LDS -> ds_load_b128
  v8h lo = *(const v8h*)(p);
  v8h hi = *(const v8h*)(p + 16);
  v16h r;
#pragma unroll
  for (int i = 0; i < 8; ++i) { r[i] = lo[i]; r[i + 8] = hi[i]; }
  return r;
}

__device__ __forceinline__ v16h load_frag_g(gwptr p) {           // global_load_b128
  v8h lo = *(gv8p)(p);
  v8h hi = *(gv8p)(p + 16);
  v16h r;
#pragma unroll
  for (int i = 0; i < 8; ++i) { r[i] = lo[i]; r[i + 8] = hi[i]; }
  return r;
}

__device__ __forceinline__ v8f wmma16(v16h a, v16h b, v8f c) {
  return __builtin_amdgcn_wmma_f32_16x16x32_f16(false, a, false, b, (short)0, c, false, false);
}

__device__ __forceinline__ v8f zero8() {
  v8f z;
#pragma unroll
  for (int i = 0; i < 8; ++i) z[i] = 0.0f;
  return z;
}

__device__ __forceinline__ float sigm(float x) { return 1.0f / (1.0f + __expf(-x)); }

// ---------------- weight convert / pad to f16 ----------------
__global__ void wconv_kernel(const float* __restrict__ Wih0, const float* __restrict__ Whh0,
                             const float* __restrict__ Wih1, const float* __restrict__ Whh1,
                             const float* __restrict__ W1,   const float* __restrict__ W2,
                             half_t* __restrict__ ws) {
  int i = blockIdx.x * 256 + threadIdx.x;
  if (i >= WS_HALFS) return;
  float v = 0.0f;
  if (i < OFF_WH0) {                       // [512][32] from [512][24]
    int r = i >> 5, c = i & 31;
    if (c < 24) v = Wih0[r * 24 + c];
  } else if (i < OFF_WI1) {
    v = Whh0[i - OFF_WH0];
  } else if (i < OFF_WH1) {
    v = Wih1[i - OFF_WI1];
  } else if (i < OFF_MW1) {
    v = Whh1[i - OFF_WH1];
  } else if (i < OFF_MW2) {                // [128][192] from [128][184]
    int j = i - OFF_MW1; int r = j / 192, c = j % 192;
    if (c < 184) v = W1[r * 184 + c];
  } else {
    v = W2[i - OFF_MW2];
  }
  ws[i] = (half_t)v;
}

// ---------------- main fused kernel ----------------
__global__ __launch_bounds__(256)
void s2s_ardiff_kernel(const float* __restrict__ x_hist, const float* __restrict__ x_future,
                       const float* __restrict__ y0,     const int* __restrict__ turb_idx,
                       const float* __restrict__ init_noise, const float* __restrict__ turb_emb,
                       const float* __restrict__ b_ih0, const float* __restrict__ b_hh0,
                       const float* __restrict__ b_ih1, const float* __restrict__ b_hh1,
                       const float* __restrict__ mb1g,  const float* __restrict__ mb2g,
                       const float* __restrict__ W3,    const float* __restrict__ b3,
                       const half_t* __restrict__ wsh,  float* __restrict__ out)
{
  __shared__ __align__(16) half_t xh[16][32];        // LSTM0 input (x_hist|emb|pad)
  __shared__ __align__(16) half_t h0[2][16][HID];    // ping-pong hidden L0
  __shared__ __align__(16) half_t h1[2][16][HID];    // ping-pong hidden L1
  __shared__ __align__(16) half_t Xs[16][192];       // MLP input (cond|y|t_e|h_e|pad)
  __shared__ __align__(16) half_t Ha[16][HID];
  __shared__ __align__(16) half_t Hb[16][HID];
  __shared__ float bias0[512];
  __shared__ float bias1[512];
  __shared__ float mb1[HID];
  __shared__ float mb2[HID];
  __shared__ float abar[TSTEPS];
  __shared__ float sfreq[8];
  __shared__ float yv[16];
  __shared__ float prev_y[16];

  const int tid  = threadIdx.x;
  const int w    = tid >> 5;     // wave id: owns output cols [w*16, w*16+16)
  const int lane = tid & 31;
  const int mrow = lane & 15;    // A-fragment row / B-fragment col
  const int hs   = lane >> 4;    // K-half select
  const int nc   = w * 16 + mrow;
  const int r0   = blockIdx.x * 16;

  // ---- init ----
  for (int i = tid; i < 512; i += 256) { bias0[i] = b_ih0[i] + b_hh0[i]; bias1[i] = b_ih1[i] + b_hh1[i]; }
  for (int i = tid; i < HID; i += 256) { mb1[i] = mb1g[i]; mb2[i] = mb2g[i]; }
  if (tid == 0) {
    float ab = 1.0f;
    for (int t = 0; t < TSTEPS; ++t) {
      float beta = 1e-4f + (0.02f - 1e-4f) * (float)t / 99.0f;
      ab *= (1.0f - beta);
      abar[t] = ab;
    }
  }
  if (tid < 8) sfreq[tid] = __expf(-9.2103403719762f * (float)tid / 8.0f);
  for (int i = tid; i < 16 * HID; i += 256) {
    int r = i >> 7, c = i & 127;
    h0[0][r][c] = (half_t)0.0f; h1[0][r][c] = (half_t)0.0f;
  }
  {                                        // turbine embedding -> xh cols 8..23
    int r = tid >> 4, c = tid & 15;
    xh[r][8 + c] = (half_t)turb_emb[turb_idx[r0 + r] * EMBD + c];
  }
  if (tid < 128) { int r = tid >> 3, c = tid & 7; xh[r][24 + c] = (half_t)0.0f; }
  for (int off = tid * 64; off < WS_HALFS; off += 256 * 64)
    __builtin_prefetch(wsh + off, 0, 1);   // global_prefetch_b8, warm L2
  __syncthreads();

  float c0r[8], c1r[8];
#pragma unroll
  for (int v = 0; v < 8; ++v) { c0r[v] = 0.0f; c1r[v] = 0.0f; }

  // weight base kept in an SGPR pair; asm barrier stops LICM from hoisting
  // ~50 B-fragments (~400 VGPRs) out of the sequential loops.
  unsigned long long wbase = (unsigned long long)wsh;

  // ================= encoder: 2-layer LSTM over 96 steps =================
  int p = 0;
  for (int t = 0; t < T_IN; ++t) {
    asm volatile("" : "+s"(wbase));
    gwptr W0p  = (gwptr)wbase + OFF_W0;
    gwptr Wh0p = (gwptr)wbase + OFF_WH0;
    gwptr Wi1p = (gwptr)wbase + OFF_WI1;
    gwptr Wh1p = (gwptr)wbase + OFF_WH1;

    if (tid < 128) {
      int r = tid >> 3, c = tid & 7;
      xh[r][c] = (half_t)x_hist[(size_t)(r0 + r) * (T_IN * F_IN) + t * F_IN + c];
    }
    __syncthreads();

    // ---- LSTM layer 0: gates = xh @ Wih0^T + h0 @ Whh0^T ----
    v8f a0 = zero8(), a1 = zero8(), a2 = zero8(), a3 = zero8();
    {
      int k0 = hs * 8;
      v16h a = load_frag_s(&xh[mrow][0] + k0);
      a0 = wmma16(a, load_frag_g(W0p + (0 * HID + nc) * 32 + k0), a0);
      a1 = wmma16(a, load_frag_g(W0p + (1 * HID + nc) * 32 + k0), a1);
      a2 = wmma16(a, load_frag_g(W0p + (2 * HID + nc) * 32 + k0), a2);
      a3 = wmma16(a, load_frag_g(W0p + (3 * HID + nc) * 32 + k0), a3);
    }
#pragma unroll
    for (int kt = 0; kt < 4; ++kt) {
      int k0 = kt * 32 + hs * 8;
      v16h a = load_frag_s(&h0[p][mrow][0] + k0);
      a0 = wmma16(a, load_frag_g(Wh0p + (0 * HID + nc) * HID + k0), a0);
      a1 = wmma16(a, load_frag_g(Wh0p + (1 * HID + nc) * HID + k0), a1);
      a2 = wmma16(a, load_frag_g(Wh0p + (2 * HID + nc) * HID + k0), a2);
      a3 = wmma16(a, load_frag_g(Wh0p + (3 * HID + nc) * HID + k0), a3);
    }
#pragma unroll
    for (int v = 0; v < 8; ++v) {
      float iv = sigm(a0[v] + bias0[nc]);
      float fv = sigm(a1[v] + bias0[HID + nc]);
      float gv = tanhf(a2[v] + bias0[2 * HID + nc]);
      float ov = sigm(a3[v] + bias0[3 * HID + nc]);
      c0r[v] = fv * c0r[v] + iv * gv;
      h0[p ^ 1][v + 8 * hs][nc] = (half_t)(ov * tanhf(c0r[v]));
    }
    __syncthreads();

    // ---- LSTM layer 1: input = new h0, state = h1 ----
    a0 = zero8(); a1 = zero8(); a2 = zero8(); a3 = zero8();
#pragma unroll
    for (int kt = 0; kt < 4; ++kt) {
      int k0 = kt * 32 + hs * 8;
      v16h a = load_frag_s(&h0[p ^ 1][mrow][0] + k0);
      a0 = wmma16(a, load_frag_g(Wi1p + (0 * HID + nc) * HID + k0), a0);
      a1 = wmma16(a, load_frag_g(Wi1p + (1 * HID + nc) * HID + k0), a1);
      a2 = wmma16(a, load_frag_g(Wi1p + (2 * HID + nc) * HID + k0), a2);
      a3 = wmma16(a, load_frag_g(Wi1p + (3 * HID + nc) * HID + k0), a3);
    }
#pragma unroll
    for (int kt = 0; kt < 4; ++kt) {
      int k0 = kt * 32 + hs * 8;
      v16h a = load_frag_s(&h1[p][mrow][0] + k0);
      a0 = wmma16(a, load_frag_g(Wh1p + (0 * HID + nc) * HID + k0), a0);
      a1 = wmma16(a, load_frag_g(Wh1p + (1 * HID + nc) * HID + k0), a1);
      a2 = wmma16(a, load_frag_g(Wh1p + (2 * HID + nc) * HID + k0), a2);
      a3 = wmma16(a, load_frag_g(Wh1p + (3 * HID + nc) * HID + k0), a3);
    }
#pragma unroll
    for (int v = 0; v < 8; ++v) {
      float iv = sigm(a0[v] + bias1[nc]);
      float fv = sigm(a1[v] + bias1[HID + nc]);
      float gv = tanhf(a2[v] + bias1[2 * HID + nc]);
      float ov = sigm(a3[v] + bias1[3 * HID + nc]);
      c1r[v] = fv * c1r[v] + iv * gv;
      h1[p ^ 1][v + 8 * hs][nc] = (half_t)(ov * tanhf(c1r[v]));
    }
    p ^= 1;
    __syncthreads();
  }

  // ================= decoder: AR diffusion =================
  // Xs columns: [0..127]=enc_out, [128..133]=exo, [134..149]=emb, [150]=prev_y,
  //             [151]=y, [152..167]=t_emb, [168..183]=h_emb, [184..191]=pad
  for (int i = tid; i < 16 * HID; i += 256) {
    int r = i >> 7, c = i & 127;
    Xs[r][c] = h1[p][r][c];
  }
  { int r = tid >> 4, c = tid & 15; Xs[r][134 + c] = xh[r][8 + c]; }
  if (tid < 128) { int r = tid >> 3, c = tid & 7; Xs[r][184 + c] = (half_t)0.0f; }
  if (tid < 16) prev_y[tid] = y0[r0 + tid];
  __syncthreads();

  for (int s = 0; s < PRED; ++s) {
    if (tid < 96) {
      int r = tid / 6, j = tid % 6;
      Xs[r][128 + j] = (half_t)x_future[(size_t)(r0 + r) * (PRED * EXO) + s * EXO + j];
    }
    if (tid < 16) {
      Xs[tid][150] = (half_t)prev_y[tid];
      yv[tid] = init_noise[s * NB + (r0 + tid)];
    }
    {   // h_e = sin_emb(s): same for all 16 rows
      int r = tid >> 4, i = tid & 15;
      float ang = (float)s * sfreq[i & 7];
      Xs[r][168 + i] = (half_t)((i < 8) ? __cosf(ang) : __sinf(ang));
    }
    __syncthreads();

    for (int k = 0; k < TSTEPS; ++k) {
      int t = TSTEPS - 1 - k;
      asm volatile("" : "+s"(wbase));
      gwptr MW1p = (gwptr)wbase + OFF_MW1;
      gwptr MW2p = (gwptr)wbase + OFF_MW2;

      {   // t_e = sin_emb(t)
        int r = tid >> 4, i = tid & 15;
        float ang = (float)t * sfreq[i & 7];
        Xs[r][152 + i] = (half_t)((i < 8) ? __cosf(ang) : __sinf(ang));
      }
      if (tid < 16) Xs[tid][151] = (half_t)yv[tid];
      __syncthreads();

      // layer1: 192(K) -> 128, relu
      v8f acc = zero8();
#pragma unroll
      for (int kt = 0; kt < 6; ++kt) {
        int k0 = kt * 32 + hs * 8;
        v16h a = load_frag_s(&Xs[mrow][0] + k0);
        acc = wmma16(a, load_frag_g(MW1p + nc * 192 + k0), acc);
      }
#pragma unroll
      for (int v = 0; v < 8; ++v) {
        float hv = acc[v] + mb1[nc];
        Ha[v + 8 * hs][nc] = (half_t)(hv > 0.0f ? hv : 0.0f);
      }
      __syncthreads();

      // layer2: 128 -> 128, relu
      acc = zero8();
#pragma unroll
      for (int kt = 0; kt < 4; ++kt) {
        int k0 = kt * 32 + hs * 8;
        v16h a = load_frag_s(&Ha[mrow][0] + k0);
        acc = wmma16(a, load_frag_g(MW2p + nc * HID + k0), acc);
      }
#pragma unroll
      for (int v = 0; v < 8; ++v) {
        float hv = acc[v] + mb2[nc];
        Hb[v + 8 * hs][nc] = (half_t)(hv > 0.0f ? hv : 0.0f);
      }
      __syncthreads();

      // layer3 (128 -> 1) + deterministic DDPM update
      if (tid < 16) {
        float e = b3[0];
        for (int hh = 0; hh < HID; ++hh)
          e += (float)Hb[tid][hh] * W3[hh];
        float beta = 1e-4f + (0.02f - 1e-4f) * (float)t / 99.0f;
        float at   = 1.0f - beta;
        float ab   = abar[t];
        float y    = yv[tid];
        float yn;
        if (t == 0) yn = (y - sqrtf(1.0f - ab) * e) / (sqrtf(ab) + 1e-8f);
        else        yn = (y - beta / (sqrtf(1.0f - ab) + 1e-8f) * e) / (sqrtf(at) + 1e-8f);
        yv[tid] = yn;
      }
      __syncthreads();
    }

    if (tid < 16) {
      prev_y[tid] = yv[tid];
      out[(size_t)(r0 + tid) * PRED + s] = yv[tid];
    }
    __syncthreads();
  }
}

extern "C" void kernel_launch(void* const* d_in, const int* in_sizes, int n_in,
                              void* d_out, int out_size, void* d_ws, size_t ws_size,
                              hipStream_t stream) {
  const float* x_hist     = (const float*)d_in[0];
  const float* x_future   = (const float*)d_in[1];
  const float* y0         = (const float*)d_in[2];
  const int*   turb_idx   = (const int*)d_in[3];
  /* d_in[4] = pred_steps (fixed = 8) */
  const float* init_noise = (const float*)d_in[5];
  const float* turb_emb   = (const float*)d_in[6];
  const float* W_ih0      = (const float*)d_in[7];
  const float* W_hh0      = (const float*)d_in[8];
  const float* b_ih0      = (const float*)d_in[9];
  const float* b_hh0      = (const float*)d_in[10];
  const float* W_ih1      = (const float*)d_in[11];
  const float* W_hh1      = (const float*)d_in[12];
  const float* b_ih1      = (const float*)d_in[13];
  const float* b_hh1      = (const float*)d_in[14];
  const float* W1         = (const float*)d_in[15];
  const float* b1         = (const float*)d_in[16];
  const float* W2         = (const float*)d_in[17];
  const float* b2         = (const float*)d_in[18];
  const float* W3         = (const float*)d_in[19];
  const float* b3         = (const float*)d_in[20];

  half_t* wsh = (half_t*)d_ws;
  float*  outp = (float*)d_out;

  wconv_kernel<<<(WS_HALFS + 255) / 256, 256, 0, stream>>>(W_ih0, W_hh0, W_ih1, W_hh1, W1, W2, wsh);
  s2s_ardiff_kernel<<<NB / 16, 256, 0, stream>>>(x_hist, x_future, y0, turb_idx, init_noise,
                                                 turb_emb, b_ih0, b_hh0, b_ih1, b_hh1,
                                                 b1, b2, W3, b3, wsh, outp);
}
